// BiRNN_22935125360876
// MI455X (gfx1250) — matrix operand order repaired and checked by reading, not compiled, not run on hardware
//
#include <hip/hip_runtime.h>
#include <hip/hip_bf16.h>

// BiRNN on MI455X (gfx1250, wave32).
//  Kernel 1: x_proj = X @ W_xh + b  (f16 WMMA, f32 accum) -> written into d_out in place.
//  Kernel 2: 512 sequential steps  h_t = tanh(xp_t + h_{t-1} @ W_hh), both directions.
//            W_hh register-resident as WMMA B-fragments; h ping-pongs via L2, staged
//            into LDS with global_load_async_to_lds_b128; one device-scope split
//            barrier per step.

typedef __attribute__((ext_vector_type(16))) _Float16 v16h;
typedef __attribute__((ext_vector_type(8)))  float    v8f;

#define T_DIM 512
#define B_DIM 64
#define D_DIM 512
#define H_DIM 512
#define LDC   1024   // 2*H, row stride of d_out's [T,B,2H] block

// Scheduling fence: VALU/SALU/WMMA may cross, DS/VMEM may not. Bounds the
// scheduler's load-prefetch window so the 128-VGPR weight file never spills.
#define SCHED_FENCE() __builtin_amdgcn_sched_barrier(0x2 | 0x4 | 0x8)

// ---------------------------------------------------------------------------
// WMMA helpers
// ---------------------------------------------------------------------------
__device__ __forceinline__ v8f wmma_f16(v16h a, v16h b, v8f c) {
  // 8 args: (neg_a, A, neg_b, B, c_mod, C, reuse_a, reuse_b)
  return __builtin_amdgcn_wmma_f32_16x16x32_f16(false, a, false, b, (short)0, c,
                                                false, false);
}

// A-matrix fragment (16x32 f16). `prow` is the lane's row pointer into the f16
// tile (row stride 512), `koff` the fragment's K offset. All index math is
// compile-time-constant relative to prow -> ds_load immediate offsets.
// Layout: lanes 0-15 row M=lane, VGPR v -> K=2v,2v+1; hi lanes pass koff+16
// (contiguous K order, matching load_b_frags).
__device__ __forceinline__ v16h load_a_frag(const _Float16* __restrict__ prow,
                                            int koff) {
  v16h a;
#pragma unroll
  for (int v = 0; v < 8; ++v) {
    const int k0 = koff + (v << 1);
    a[2 * v]     = prow[k0];
    a[2 * v + 1] = prow[k0 + 1];
  }
  return a;
}

// B-matrix fragments (32x16 f16 per K-chunk) for one 16-column strip of a
// [512 x 512] row-major f32 weight matrix, converted to f16, packed in contiguous K
// order: lanes 0-15 col N=lane, K=2v,2v+1; lanes 16-31 K+16.
__device__ __forceinline__ void load_b_frags(const float* __restrict__ W,
                                             int ncol_global, int lane,
                                             v16h bfrag[16]) {
  const int n     = ncol_global + (lane & 15);
  const int khalf = (lane >> 4) << 4;   // 0 or 16
#pragma unroll 4
  for (int kc = 0; kc < 16; ++kc) {
    v16h bfv;
#pragma unroll
    for (int v = 0; v < 8; ++v) {
      const int k0 = (kc << 5) + (v << 1) + khalf;
      bfv[2 * v]     = (_Float16)W[(size_t)k0 * H_DIM + n];
      bfv[2 * v + 1] = (_Float16)W[(size_t)(k0 + 1) * H_DIM + n];
    }
    bfrag[kc] = bfv;
  }
}

// ---------------------------------------------------------------------------
// Workspace init (harness poisons d_ws; barrier counters + h ping-pong must be
// zeroed every call so graph replays are deterministic).
// ---------------------------------------------------------------------------
__global__ void init_ws_kernel(int* __restrict__ ws, int n) {
  const int i = blockIdx.x * blockDim.x + threadIdx.x;
  if (i < n) ws[i] = 0;
}

// ---------------------------------------------------------------------------
// Kernel 1: x_proj for both directions, written to d_out[t*B+b, dir*H + n].
// Block: 256 thr (8 waves). Wave w owns a 16-col strip of a 128-col block
// column; B-frags (W_xh) register-resident; X rows staged in LDS as f16.
// Grid: (64, 4, 2); each block loops over 8 chunks of 64 rows (M = T*B = 32768).
// ---------------------------------------------------------------------------
__global__ void __launch_bounds__(256, 1)
__attribute__((amdgpu_waves_per_eu(1))) xproj_kernel(
    const float* __restrict__ X,
    const float* __restrict__ Wf, const float* __restrict__ Wb,
    const float* __restrict__ bf, const float* __restrict__ bb,
    float* __restrict__ out) {
  extern __shared__ _Float16 Ash[];          // 64 x 512 f16 = 64 KB
  const int dir  = blockIdx.z;
  const float* W    = dir ? Wb : Wf;
  const float* bias = dir ? bb : bf;
  const int n0   = blockIdx.y * 128;
  const int w    = threadIdx.x >> 5;
  const int lane = threadIdx.x & 31;
  const int nw   = n0 + (w << 4);
  const int ncol = lane & 15;
  const int mbase = (lane >> 4) << 3;

  v16h bfrag[16];
  load_b_frags(W, nw, lane, bfrag);
  const float bv = bias[nw + ncol];

  // Per-lane invariants with pure-int offsets.
  const _Float16* aprow = Ash + (lane & 15) * 512;           // A-frag row ptr
  // lane base for C stores (row mbase, col dir*H + nw + ncol), advanced per mc
  float* cbase0 = out + (size_t)mbase * LDC + (size_t)(dir * H_DIM + nw + ncol);

  for (int mc = blockIdx.x; mc < 512; mc += gridDim.x) {
    const int m0 = mc * 64;
    const float* __restrict__ xbase = X + (size_t)m0 * D_DIM;   // uniform base
    if (mc + (int)gridDim.x < 512)
      __builtin_prefetch(X + (size_t)(mc + gridDim.x) * 64 * D_DIM, 0, 0);
    __syncthreads();  // WAR on Ash vs previous iteration
    // stage 64 rows of X (f32 -> f16) into LDS, fully coalesced:
    // one global_load_b128 + one ds_store_b64 per 4 elements.
    for (unsigned i = threadIdx.x; i < 64u * 128u; i += 256u) {
      const unsigned off = i << 2;             // element offset (x4 floats)
      const float4 xv = *(const float4*)(xbase + off);
      union { _Float16 h[4]; uint2 u; } pk;
      pk.h[0] = (_Float16)xv.x; pk.h[1] = (_Float16)xv.y;
      pk.h[2] = (_Float16)xv.z; pk.h[3] = (_Float16)xv.w;
      *(uint2*)(Ash + off) = pk.u;
    }
    __syncthreads();

    v8f c[4];
#pragma unroll
    for (int mt = 0; mt < 4; ++mt)
#pragma unroll
      for (int r = 0; r < 8; ++r) c[mt][r] = bv;

#pragma unroll
    for (int kc = 0; kc < 16; ++kc) {
#pragma unroll
      for (int mt = 0; mt < 4; ++mt) {
        const v16h a = load_a_frag(aprow + mt * 16 * 512, kc * 32 + ((lane >> 4) << 4));
        c[mt] = wmma_f16(a, bfrag[kc], c[mt]);
      }
      SCHED_FENCE();   // keep <=1 iteration of A-frags in flight
    }

    float* __restrict__ cb = cbase0 + (size_t)m0 * LDC;
#pragma unroll
    for (int mt = 0; mt < 4; ++mt)
#pragma unroll
      for (int r = 0; r < 8; ++r)
        cb[(mt * 16 + r) * LDC] = c[mt][r];      // immediate-offset stores
  }
}

// ---------------------------------------------------------------------------
// Kernel 2: the recurrence. 8 blocks = 2 dirs x 4 column slices of 128.
// Per step: async-stage h(t-1) (64KB f16) from the L2 ping-pong buffer into
// LDS (global_load_async_to_lds_b128, ASYNCcnt-tracked, no VGPR round trip),
// 64 WMMAs/wave with register-resident W_hh B-frags, tanh, write slice back
// (f32 in-place over x_proj in d_out; f16 into the other parity buffer),
// then one device-scope split barrier per direction.
// ---------------------------------------------------------------------------
__global__ void __launch_bounds__(256, 1)
__attribute__((amdgpu_waves_per_eu(1))) rnn_step_kernel(
    const float* __restrict__ Whh_f, const float* __restrict__ Whh_b,
    float* __restrict__ out, _Float16* __restrict__ hbuf,
    int* __restrict__ counters) {
  extern __shared__ _Float16 hsh[];          // 64 x 512 f16 = 64 KB
  const int dir   = blockIdx.x >> 2;
  const int slice = blockIdx.x & 3;
  const int n0    = slice * 128;
  const float* W  = dir ? Whh_b : Whh_f;
  const int w     = threadIdx.x >> 5;
  const int lane  = threadIdx.x & 31;
  const int nw    = n0 + (w << 4);
  const int ncol  = lane & 15;
  const int mbase = (lane >> 4) << 3;
  const int ncg   = nw + ncol;               // global column in H

  v16h bfrag[16];                            // 512x16 strip of W_hh: 128 VGPRs
  load_b_frags(W, nw, lane, bfrag);

  // ---- per-lane invariant bases (pure-int offsets inside the loop) ---------
  // Parity buffers are a fixed 2*B*H element stride apart; keep ONE base per
  // stream and add parity*stride per step (saves live 64-bit pointer pairs).
  const size_t PAR_STRIDE_H = (size_t)2 * B_DIM * H_DIM;          // halves
  const _Float16* aprow = hsh + (lane & 15) * 512;      // A-frag row ptr
  // d_out lane base at row (b = mbase), col dir*H + ncg; step adds tt*B rows.
  float* const out_lane = out + (size_t)mbase * LDC + (size_t)(dir * H_DIM + ncg);
  // h ping-pong lane base (parity 0); parity 1 = +PAR_STRIDE_H
  _Float16* const hd_base = hbuf + (size_t)dir * B_DIM * H_DIM +
                            (size_t)mbase * H_DIM + ncg;
  // async staging base (parity 0); LDS AS3 offset == low 32 bits of generic ptr
  const unsigned lds_base = (unsigned)(uintptr_t)hsh + (unsigned)threadIdx.x * 16u;
  const char* const gsrc_base =
      (const char*)(hbuf + (size_t)dir * B_DIM * H_DIM) + (size_t)threadIdx.x * 16u;

  for (int t = 0; t < T_DIM; ++t) {
    const int tt = dir ? (T_DIM - 1 - t) : t;

    // ---- async stage h(t-1) from parity buffer ((t-1)&1)==((t+1)&1) --------
    {
      const char* gsrc = gsrc_base + (size_t)((t + 1) & 1) * (PAR_STRIDE_H * 2);
      // 64 KB = 4096 x 16B; 256 threads x 16 ops, 16B per lane per op.
#pragma unroll
      for (int i = 0; i < 16; ++i) {
        const unsigned      ldsa = lds_base + (unsigned)i * 4096u;
        const unsigned long gva  = (unsigned long)(uintptr_t)gsrc +
                                   (unsigned long)i * 4096ul;
        asm volatile("global_load_async_to_lds_b128 %0, %1, off"
                     :
                     : "v"(ldsa), "v"(gva)
                     : "memory");
      }
      asm volatile("s_wait_asynccnt 0x0" ::: "memory");
    }
    __syncthreads();

    const float* __restrict__ xp = out_lane + (size_t)tt * (B_DIM * LDC);
    v8f c[4];
#pragma unroll
    for (int mt = 0; mt < 4; ++mt)
#pragma unroll
      for (int r = 0; r < 8; ++r)
        c[mt][r] = xp[(mt * 16 + r) * LDC];        // immediate-offset loads

#pragma unroll
    for (int kc = 0; kc < 16; ++kc) {
#pragma unroll
      for (int mt = 0; mt < 4; ++mt) {
        const v16h a = load_a_frag(aprow + mt * 16 * 512, kc * 32 + ((lane >> 4) << 4));
        c[mt] = wmma_f16(a, bfrag[kc], c[mt]);
      }
      SCHED_FENCE();   // keep <=1 iteration of A-frags in flight
    }

    float* __restrict__ od = out_lane + (size_t)tt * (B_DIM * LDC);
    _Float16* __restrict__ hdst = hd_base + (size_t)(t & 1) * PAR_STRIDE_H;
#pragma unroll
    for (int mt = 0; mt < 4; ++mt)
#pragma unroll
      for (int r = 0; r < 8; ++r) {
        const float v = tanhf(c[mt][r]);
        c[mt][r] = v;
        od[(mt * 16 + r) * LDC] = v;               // immediate-offset stores
        hdst[(mt * 16 + r) * H_DIM] = (_Float16)v;
      }
    if (t == T_DIM - 1) {                          // f_H / b_H tail, once
      float* __restrict__ hN_lane =
          out + (size_t)T_DIM * B_DIM * LDC + (size_t)dir * B_DIM * H_DIM +
          (size_t)mbase * H_DIM + ncg;
#pragma unroll
      for (int mt = 0; mt < 4; ++mt)
#pragma unroll
        for (int r = 0; r < 8; ++r)
          hN_lane[(mt * 16 + r) * H_DIM] = c[mt][r];
    }

    // ---- device-scope split barrier across the 4 blocks of this direction ----
    __threadfence();          // make this thread's global writes device-visible
    __syncthreads();          // all waves of the WG done
    if (threadIdx.x == 0) {
      __hip_atomic_fetch_add(&counters[dir], 1, __ATOMIC_RELEASE,
                             __HIP_MEMORY_SCOPE_AGENT);
      const int target = 4 * (t + 1);
      while (__hip_atomic_load(&counters[dir], __ATOMIC_ACQUIRE,
                               __HIP_MEMORY_SCOPE_AGENT) < target) {
        __builtin_amdgcn_s_sleep(2);
      }
    }
    __syncthreads();
    __threadfence();          // acquire: invalidate WGP caches before re-reading h
  }
}

// ---------------------------------------------------------------------------
// Host launch
// ---------------------------------------------------------------------------
extern "C" void kernel_launch(void* const* d_in, const int* in_sizes, int n_in,
                              void* d_out, int out_size, void* d_ws,
                              size_t ws_size, hipStream_t stream) {
  const float* X     = (const float*)d_in[0];  // [T,B,D]
  const float* Wxh_f = (const float*)d_in[1];  // [D,H]
  const float* Whh_f = (const float*)d_in[2];  // [H,H]
  const float* bh_f  = (const float*)d_in[3];  // [H]
  const float* Wxh_b = (const float*)d_in[4];
  const float* Whh_b = (const float*)d_in[5];
  const float* bh_b  = (const float*)d_in[6];
  float* out = (float*)d_out;

  // ws layout: h ping-pong [2 parity][2 dir][B][H] f16 (256 KB), then counters.
  const size_t HBUF_BYTES = (size_t)2 * 2 * B_DIM * H_DIM * sizeof(_Float16);
  _Float16* hbuf = (_Float16*)d_ws;
  int* counters  = (int*)((char*)d_ws + HBUF_BYTES);

  const int init_ints = (int)(HBUF_BYTES / sizeof(int)) + 2;
  init_ws_kernel<<<(init_ints + 255) / 256, 256, 0, stream>>>((int*)d_ws,
                                                              init_ints);

  const size_t lds_bytes = (size_t)B_DIM * 512 * sizeof(_Float16);  // 64 KB
  dim3 g1(64, 4, 2);
  xproj_kernel<<<g1, 256, lds_bytes, stream>>>(X, Wxh_f, Wxh_b, bh_f, bh_b, out);

  rnn_step_kernel<<<8, 256, lds_bytes, stream>>>(Whh_f, Whh_b, out, hbuf,
                                                 counters);
}
